// GCN_82179904241993
// MI455X (gfx1250) — compile-verified, matrix operand order checked by reading, not run on hardware
//
#include <hip/hip_runtime.h>
#include <hip/hip_bf16.h>

typedef __attribute__((ext_vector_type(16))) __bf16 v16bf;
typedef __attribute__((ext_vector_type(8)))  float  v8f;

constexpr int N_NODES = 50000;
constexpr int N_EDGES = 800000;
constexpr int NFEAT   = 512;
constexpr int NHID    = 128;
constexpr int NCLASS  = 64;

// ---- CDNA5 async global->LDS path (probe via __has_builtin; fallback is plain HIP)
#if defined(__gfx1250__) && __has_builtin(__builtin_amdgcn_global_load_async_to_lds_b32) && \
    __has_builtin(__builtin_amdgcn_s_wait_asynccnt)
#define HAS_ASYNC_LDS 1
typedef __attribute__((address_space(1))) int gbl_i32;   // clang prints this as '__device__ int *'
typedef __attribute__((address_space(3))) int lds_i32;   // clang prints this as '__shared__ int *'
#else
#define HAS_ASYNC_LDS 0
#endif

// ---------- bf16 helpers (bit-exact, no reliance on __bf16 arithmetic) ----------
__device__ __forceinline__ __bf16 f32_to_bf16_rte(float f) {
    unsigned u = __float_as_uint(f);
    unsigned r = (u + 0x7FFFu + ((u >> 16) & 1u)) >> 16;   // round-to-nearest-even
    unsigned short s = (unsigned short)r;
    __bf16 b;
    __builtin_memcpy(&b, &s, 2);
    return b;
}
__device__ __forceinline__ float bf16_to_f32(__bf16 b) {
    unsigned short s;
    __builtin_memcpy(&s, &b, 2);
    return __uint_as_float(((unsigned)s) << 16);
}
// Split f into bf16 hi/lo, returning the two 16-bit patterns.
__device__ __forceinline__ void bf16_split_bits(float f, unsigned short& hb, unsigned short& lb) {
    __bf16 h = f32_to_bf16_rte(f);
    __bf16 l = f32_to_bf16_rte(f - bf16_to_f32(h));
    __builtin_memcpy(&hb, &h, 2);
    __builtin_memcpy(&lb, &l, 2);
}

// ---------------------------------------------------------------------------
// Pack a row-major K x N fp32 weight matrix into WMMA B-fragment order,
// split into bf16 hi/lo.  Fragment layout (CDNA5 ISA 7.12.2, 16-bit B, 32x16):
// lane l: N = l&15, kBase = (l>>4)*8; element e -> K = kc*32 + kBase + (e<8 ? e : e+8)
// Packed index: ((kc*NT + nt)*32 + lane)*16 + e      (NT = N/16)
// ---------------------------------------------------------------------------
__global__ void gcn_pack_b(const float* __restrict__ W, int K, int N,
                           __bf16* __restrict__ hi, __bf16* __restrict__ lo) {
    int idx = blockIdx.x * blockDim.x + threadIdx.x;
    if (idx >= K * N) return;
    int NT     = N >> 4;
    int e      = idx & 15;
    int fragId = idx >> 4;
    int lane   = fragId & 31;
    int g      = fragId >> 5;
    int nt     = g % NT;
    int kc     = g / NT;
    int kBase  = (lane >> 4) << 3;
    int k      = (kc << 5) + kBase + (e < 8 ? e : e + 8);
    int n      = (nt << 4) + (lane & 15);
    unsigned short hb, lb;
    bf16_split_bits(W[k * N + n], hb, lb);
    __builtin_memcpy(&hi[idx], &hb, 2);
    __builtin_memcpy(&lo[idx], &lb, 2);
}

// ---------------------------------------------------------------------------
// D = A(MxK fp32) * B(KxN, pre-packed bf16 hi/lo) [+ bias], via bf16x3 split
// WMMA (v_wmma_f32_16x16x32_bf16, f32 accumulate).
// Block = NT waves (NT = N/16): one 16-row M-tile, full N width.
// A fragments are converted ONCE per block per k-chunk, cooperatively, and
// stored in fragment order in LDS (2 bf16 per dword store); each wave then
// grabs them with two aligned ds_load_b128.
// ---------------------------------------------------------------------------
__global__ void gcn_gemm_bf16x3(const float* __restrict__ A,
                                const __bf16* __restrict__ Bhi,
                                const __bf16* __restrict__ Blo,
                                const float* __restrict__ bias,  // length N or nullptr
                                float* __restrict__ D,
                                int K, int N) {
    const int NT   = N >> 4;
    const int KC   = K >> 5;
    const int m0   = blockIdx.x << 4;
    const int tid  = threadIdx.x;
    const int wave = tid >> 5;        // == nt
    const int lane = tid & 31;

    __shared__ alignas(32) __bf16 aHi[32 * 16];   // A fragment, hi part
    __shared__ alignas(32) __bf16 aLo[32 * 16];   // A fragment, lo part

    const v16bf* BhiV = (const v16bf*)Bhi;
    const v16bf* BloV = (const v16bf*)Blo;
    unsigned* aHiW = (unsigned*)aHi;
    unsigned* aLoW = (unsigned*)aLo;

    v8f acc = {};

    for (int kc = 0; kc < KC; ++kc) {
        __syncthreads();
        // Cooperative A-fragment conversion: 256 dword-pairs cover 32x16 elems.
        // Pair p: fragment lane = p>>3, elements e0 = (p&7)*2 and e0+1
        // (K indices are consecutive for such pairs).
        for (int p = tid; p < 256; p += blockDim.x) {
            const int flane = p >> 3;
            const int e0    = (p & 7) << 1;
            const int row   = m0 + (flane & 15);
            const int kBase = (flane >> 4) << 3;
            const int k0    = (kc << 5) + kBase + (e0 < 8 ? e0 : e0 + 8);
            const float* src = &A[row * K + k0];
            unsigned short h0, l0, h1, l1;
            bf16_split_bits(src[0], h0, l0);
            bf16_split_bits(src[1], h1, l1);
            aHiW[p] = (unsigned)h0 | ((unsigned)h1 << 16);
            aLoW[p] = (unsigned)l0 | ((unsigned)l1 << 16);
        }
        __syncthreads();

        const v16bf ahi = *(const v16bf*)&aHi[lane << 4];
        const v16bf alo = *(const v16bf*)&aLo[lane << 4];
        const int fidx = (kc * NT + wave) * 32 + lane;
        const v16bf bhi = BhiV[fidx];
        const v16bf blo = BloV[fidx];

        // acc += hi*hi + lo*hi + hi*lo   (~fp32-accurate product, f32 accum)
        acc = __builtin_amdgcn_wmma_f32_16x16x32_bf16(false, ahi, false, bhi,
                                                      (short)0, acc, false, false);
        acc = __builtin_amdgcn_wmma_f32_16x16x32_bf16(false, alo, false, bhi,
                                                      (short)0, acc, false, false);
        acc = __builtin_amdgcn_wmma_f32_16x16x32_bf16(false, ahi, false, blo,
                                                      (short)0, acc, false, false);
    }

    // C/D layout: VGPR r -> M = r (lanes 0-15) / r+8 (lanes 16-31), N = lane&15.
    const int n     = (wave << 4) + (lane & 15);
    const float bv  = bias ? bias[n] : 0.0f;
    const int rbase = m0 + ((lane >> 4) << 3);
    #pragma unroll
    for (int r = 0; r < 8; ++r) {
        D[(rbase + r) * N + n] = acc[r] + bv;
    }
}

// ---------------------------------------------------------------------------
// Segmented SpMM: out[node,:] = sum_{e in segment(node)} vals[e] * H[cols[e],:]
// rows[] is sorted; each block (128 threads = feature dim) binary-searches its
// segment and chunks cols/vals through LDS — via ASYNCcnt-tracked
// global_load_async_to_lds_b32 when the toolchain exposes it.
// Gather source H (25.6 MB) is L2-resident on MI455X (192 MB L2).
// ---------------------------------------------------------------------------
__global__ void gcn_spmm(const int* __restrict__ rows, const int* __restrict__ cols,
                         const float* __restrict__ vals, const float* __restrict__ H,
                         const float* __restrict__ bias, int do_relu,
                         float* __restrict__ out, int nEdges) {
    const int node = blockIdx.x;
    const int t    = threadIdx.x;  // 0..127 (feature dim)

    __shared__ int   seg[2];
    __shared__ int   s_col[128];
    __shared__ float s_val[128];

    if (t < 2) {
        int target = node + t;
        int lo = 0, hi = nEdges;
        while (lo < hi) {
            int mid = (lo + hi) >> 1;
            if (rows[mid] < target) lo = mid + 1; else hi = mid;
        }
        seg[t] = lo;
    }
    __syncthreads();
    const int start = seg[0];
    const int end   = seg[1];

    float acc = 0.0f;
    for (int base = start; base < end; base += 128) {
        int cnt = end - base;
        if (cnt > 128) cnt = 128;
#if HAS_ASYNC_LDS
        if (t < cnt) {
            __builtin_amdgcn_global_load_async_to_lds_b32(
                (gbl_i32*)(cols + base + t), (lds_i32*)(&s_col[t]), 0, 0);
            __builtin_amdgcn_global_load_async_to_lds_b32(
                (gbl_i32*)(vals + base + t), (lds_i32*)(&s_val[t]), 0, 0);
        }
        __builtin_amdgcn_s_wait_asynccnt(0);
        __syncthreads();
#else
        if (t < cnt) {
            s_col[t] = cols[base + t];
            s_val[t] = vals[base + t];
        }
        __syncthreads();
#endif
        for (int i = 0; i < cnt; ++i) {
            if (i + 1 < cnt)
                __builtin_prefetch(&H[s_col[i + 1] * NHID + t], 0, 0);
            acc = fmaf(s_val[i], H[s_col[i] * NHID + t], acc);
        }
        __syncthreads();
    }

    float v = acc + (bias ? bias[t] : 0.0f);
    if (do_relu) v = fmaxf(v, 0.0f);
    out[node * NHID + t] = v;
}

// ---------------------------------------------------------------------------
// Row-wise log_softmax over 64 classes; one wave32 per row (2 cols per lane).
// ---------------------------------------------------------------------------
__global__ void gcn_logsoftmax(const float* __restrict__ L, float* __restrict__ out) {
    int row  = (blockIdx.x << 3) + (threadIdx.x >> 5);
    int lane = threadIdx.x & 31;
    if (row >= N_NODES) return;
    const float* lr = L + row * NCLASS;
    float a = lr[lane], b = lr[lane + 32];
    float m = fmaxf(a, b);
    #pragma unroll
    for (int off = 16; off > 0; off >>= 1) m = fmaxf(m, __shfl_xor(m, off, 32));
    float s = __expf(a - m) + __expf(b - m);
    #pragma unroll
    for (int off = 16; off > 0; off >>= 1) s += __shfl_xor(s, off, 32);
    float lse = m + __logf(s);
    out[row * NCLASS + lane]      = a - lse;
    out[row * NCLASS + lane + 32] = b - lse;
}

// ---------------------------------------------------------------------------
extern "C" void kernel_launch(void* const* d_in, const int* in_sizes, int n_in,
                              void* d_out, int out_size, void* d_ws, size_t ws_size,
                              hipStream_t stream) {
    (void)in_sizes; (void)n_in; (void)out_size; (void)ws_size;

    const float* x    = (const float*)d_in[0];
    const int*   rows = (const int*)  d_in[1];
    const int*   cols = (const int*)  d_in[2];
    const float* vals = (const float*)d_in[3];
    const float* W1   = (const float*)d_in[4];
    const float* b1   = (const float*)d_in[5];
    const float* W2   = (const float*)d_in[6];
    const float* b2   = (const float*)d_in[7];
    float* out = (float*)d_out;

    // Workspace layout (buffers ping-pong; all offsets 256B-aligned):
    //   bufA [25.6 MB] : H1 (GEMM1 out) then H3 (SpMM2 out)
    //   bufB [25.6 MB] : H2 (SpMM1 out) then logits (GEMM2 out)
    //   packed weights [~0.3 MB]
    char*   ws   = (char*)d_ws;
    float*  bufA = (float*)(ws);
    float*  bufB = (float*)(ws + 25600000);
    __bf16* w1hi = (__bf16*)(ws + 51200000);
    __bf16* w1lo = w1hi + NFEAT * NHID;
    __bf16* w2hi = w1lo + NFEAT * NHID;
    __bf16* w2lo = w2hi + NHID * NCLASS;

    // Pack weights into WMMA fragment order (bf16 hi/lo split).
    gcn_pack_b<<<(NFEAT * NHID + 255) / 256, 256, 0, stream>>>(W1, NFEAT, NHID, w1hi, w1lo);
    gcn_pack_b<<<(NHID * NCLASS + 255) / 256, 256, 0, stream>>>(W2, NHID, NCLASS, w2hi, w2lo);

    // H1 = X @ W1                       [50000 x 128]
    gcn_gemm_bf16x3<<<N_NODES / 16, (NHID / 16) * 32, 0, stream>>>(
        x, w1hi, w1lo, nullptr, bufA, NFEAT, NHID);

    // H2 = relu(A @ H1 + b1)            [50000 x 128]
    gcn_spmm<<<N_NODES, 128, 0, stream>>>(rows, cols, vals, bufA, b1, 1, bufB, N_EDGES);

    // H3 = A @ H2                       [50000 x 128]
    gcn_spmm<<<N_NODES, 128, 0, stream>>>(rows, cols, vals, bufB, nullptr, 0, bufA, N_EDGES);

    // logits = H3 @ W2 + b2             [50000 x 64]
    gcn_gemm_bf16x3<<<N_NODES / 16, (NCLASS / 16) * 32, 0, stream>>>(
        bufA, w2hi, w2lo, b2, bufB, NHID, NCLASS);

    // out = log_softmax(logits)
    gcn_logsoftmax<<<(N_NODES + 7) / 8, 256, 0, stream>>>(bufB, out);
}
